// GATv2ActorCritic_13262859010781
// MI455X (gfx1250) — matrix operand order repaired and checked by reading, not compiled, not run on hardware
//
#include <hip/hip_runtime.h>
#include <hip/hip_bf16.h>

// ---------------------------------------------------------------------------
// GATv2 2-layer GNN forward for MI455X (gfx1250, wave32).
//   - Edge phase (dominant, ~2x550K edges x 256/64 ch) uses f16 node features
//     to halve HBM gather traffic; f32 atomics accumulate into L2-resident
//     accumulators (acc1=51MB, acc2=13MB << 192MB L2).
//   - conv2 node transform is a GEMM [50000x256]@[256x128] via
//     v_wmma_f32_16x16x32_f16; the weight matrix is pre-swizzled into the
//     per-lane B-fragment layout so fragments load as global_load_b128.
// ---------------------------------------------------------------------------

typedef __attribute__((ext_vector_type(16))) _Float16 v16h;
typedef __attribute__((ext_vector_type(8)))  _Float16 v8h;
typedef __attribute__((ext_vector_type(2)))  _Float16 v2h;
typedef __attribute__((ext_vector_type(8)))  float    v8f;

union frag16 { v16h v; v8h h[2]; };

// Order-preserving monotone map float -> uint for atomicMax on floats.
__device__ __forceinline__ unsigned encOrd(float f) {
  unsigned u = __float_as_uint(f);
  return (u & 0x80000000u) ? ~u : (u | 0x80000000u);
}
__device__ __forceinline__ float decOrd(unsigned m) {
  unsigned u = (m & 0x80000000u) ? (m ^ 0x80000000u) : ~m;
  return __uint_as_float(u);
}

// ---- conv1 helper kernels --------------------------------------------------

// degree + sum of edge_attr per destination (for self-loop fill_value='mean')
__global__ void k_deg(const int* __restrict__ eidx, const float* __restrict__ eattr,
                      float* __restrict__ deg, float* __restrict__ loopattr, int E) {
  int e = blockIdx.x * blockDim.x + threadIdx.x;
  if (e >= E) return;
  int dst = eidx[(size_t)E + e];
  atomicAdd(deg + dst, 1.0f);
  atomicAdd(loopattr + 2 * (size_t)dst,     eattr[2 * (size_t)e]);
  atomicAdd(loopattr + 2 * (size_t)dst + 1, eattr[2 * (size_t)e + 1]);
}

__global__ void k_loopdiv(float* __restrict__ loopattr, const float* __restrict__ deg, int N) {
  int n = blockIdx.x * blockDim.x + threadIdx.x;
  if (n >= N) return;
  float d = fmaxf(deg[n], 1.0f);
  loopattr[2 * (size_t)n]     /= d;
  loopattr[2 * (size_t)n + 1] /= d;
}

// xl1 = x @ W1l + b1l ; xr1 = x @ W1r + b1r  (IN_CH=2 rank-2 update), f16 out
__global__ void k_node1(const float* __restrict__ x,
                        const float* __restrict__ W1l, const float* __restrict__ b1l,
                        const float* __restrict__ W1r, const float* __restrict__ b1r,
                        _Float16* __restrict__ xl1h, _Float16* __restrict__ xr1h, int N) {
  size_t idx = (size_t)blockIdx.x * blockDim.x + threadIdx.x;
  if (idx >= (size_t)N * 256) return;
  int j = (int)(idx & 255);
  size_t n = idx >> 8;
  float x0 = x[2 * n], x1 = x[2 * n + 1];
  xl1h[idx] = (_Float16)(x0 * W1l[j] + x1 * W1l[256 + j] + b1l[j]);
  xr1h[idx] = (_Float16)(x0 * W1r[j] + x1 * W1r[256 + j] + b1r[j]);
}

// Pass 1: attention logits per (edge, head) + segment max (wave per edge,
// 8 lanes per head, fully coalesced 16B/lane gathers)
__global__ void __launch_bounds__(256) k_att1_logits(
    const int* __restrict__ eidx, const float* __restrict__ eattr,
    const float* __restrict__ loopattr,
    const _Float16* __restrict__ xl1h, const _Float16* __restrict__ xr1h,
    const float* __restrict__ W1e, const float* __restrict__ att1,
    float* __restrict__ alpha1, unsigned* __restrict__ amax1, int E, int EP) {
  int e = blockIdx.x * 8 + (threadIdx.x >> 5);
  if (e >= EP) return;
  int l = threadIdx.x & 31;
  int src, dst; float ea0, ea1;
  if (e < E) {
    src = eidx[e]; dst = eidx[(size_t)E + e];
    ea0 = eattr[2 * (size_t)e]; ea1 = eattr[2 * (size_t)e + 1];
  } else {
    int n = e - E; src = n; dst = n;
    ea0 = loopattr[2 * (size_t)n]; ea1 = loopattr[2 * (size_t)n + 1];
  }
  int c0 = l * 8;
  v8h xlv = *(const v8h*)(xl1h + (size_t)src * 256 + c0);
  v8h xrv = *(const v8h*)(xr1h + (size_t)dst * 256 + c0);
  float p = 0.0f;
#pragma unroll
  for (int i = 0; i < 8; ++i) {
    int j = c0 + i;
    float v = (float)xlv[i] + (float)xrv[i] + ea0 * W1e[j] + ea1 * W1e[256 + j];
    v = (v > 0.0f) ? v : 0.2f * v;            // leaky_relu(0.2)
    p += v * att1[j];                          // att1 flat [h*64+c] == [j]
  }
  p += __shfl_xor(p, 1, 32);
  p += __shfl_xor(p, 2, 32);
  p += __shfl_xor(p, 4, 32);
  if ((l & 7) == 0) {
    int h = c0 >> 6;
    alpha1[(size_t)e * 4 + h] = p;
    atomicMax(amax1 + (size_t)dst * 4 + h, encOrd(p));
  }
}

// Pass 2: exp(logit - max) + segment denominator
__global__ void k_att1_exp(const int* __restrict__ eidx, float* __restrict__ alpha1,
                           const unsigned* __restrict__ amax1, float* __restrict__ denom1,
                           int E, int EP) {
  int idx = blockIdx.x * blockDim.x + threadIdx.x;
  if (idx >= EP * 4) return;
  int e = idx >> 2, h = idx & 3;
  int dst = (e < E) ? eidx[(size_t)E + e] : (e - E);
  float amax = decOrd(amax1[(size_t)dst * 4 + h]);
  float ex = __expf(alpha1[idx] - amax);
  alpha1[idx] = ex;
  atomicAdd(denom1 + (size_t)dst * 4 + h, ex);
}

// Pass 3: out[dst] += xl[src] * alpha  (atomics land in L2-resident acc1)
__global__ void __launch_bounds__(256) k_att1_aggr(
    const int* __restrict__ eidx, const float* __restrict__ alpha1,
    const float* __restrict__ denom1, const _Float16* __restrict__ xl1h,
    float* __restrict__ acc1, int E, int EP) {
  int e = blockIdx.x * 8 + (threadIdx.x >> 5);
  if (e >= EP) return;
  int l = threadIdx.x & 31;
  int src, dst;
  if (e < E) { src = eidx[e]; dst = eidx[(size_t)E + e]; }
  else       { src = dst = e - E; }
  int c0 = l * 8, h = c0 >> 6;
  float w = alpha1[(size_t)e * 4 + h] / (denom1[(size_t)dst * 4 + h] + 1e-16f);
  v8h xlv = *(const v8h*)(xl1h + (size_t)src * 256 + c0);
  float* accp = acc1 + (size_t)dst * 256 + c0;
#pragma unroll
  for (int i = 0; i < 8; ++i) atomicAdd(accp + i, (float)xlv[i] * w);
}

// h = elu(acc1 + bias1), stored f16 for the WMMA GEMM
__global__ void k_finish1(const float* __restrict__ acc1, const float* __restrict__ bias1,
                          _Float16* __restrict__ h16, int N) {
  size_t idx = (size_t)blockIdx.x * blockDim.x + threadIdx.x;
  if (idx >= (size_t)N * 256) return;
  int j = (int)(idx & 255);
  float v = acc1[idx] + bias1[j];
  v = (v > 0.0f) ? v : expm1f(v);
  h16[idx] = (_Float16)v;
}

// Fuse W2l|W2r -> f16, pre-swizzled into the WMMA B-fragment layout:
//   wsw[(((ntile*8 + kstep)*32 + lane)*16) + i]
//     = B[kstep*32 + (lane&16 ? 16 : 0) + i][ntile*16 + (lane&15)]
// so each lane's fragment is 32 contiguous bytes (two b128 loads).
__global__ void k_wcat_sw(const float* __restrict__ W2l, const float* __restrict__ W2r,
                          _Float16* __restrict__ wsw) {
  int idx = blockIdx.x * blockDim.x + threadIdx.x;
  if (idx >= 8 * 8 * 32 * 16) return;
  int i  = idx & 15;
  int l  = (idx >> 4) & 31;
  int ki = (idx >> 9) & 7;
  int t  = idx >> 12;
  int col = t * 16 + (l & 15);                 // fused output column (0..127)
  int k   = ki * 32 + ((l & 16) ? 16 : 0) + i; // K index (0..255)
  float v = (col < 64) ? W2l[(size_t)k * 64 + col]
                       : W2r[(size_t)k * 64 + (col - 64)];
  wsw[idx] = (_Float16)v;
}

// ---- WMMA GEMM: [N x 256] @ [256 x 128] -> xl2 (cols 0-63) / xr2 (64-127) ---
// One wave32 per 16x16 C tile; v_wmma_f32_16x16x32_f16 over 8 K-steps.
// 16-bit A 16x32 / B 32x16 use contiguous k = 16*(lane/16) + i in both; f32 C 16x16 per ISA layout.
__global__ void __launch_bounds__(128) k_gemm2(
    const _Float16* __restrict__ h16, const _Float16* __restrict__ wsw,
    const float* __restrict__ b2l, const float* __restrict__ b2r,
    _Float16* __restrict__ xl2h, _Float16* __restrict__ xr2h, int N, int mtiles) {
  int wave = blockIdx.x * 4 + (threadIdx.x >> 5);
  int l = threadIdx.x & 31;
  int mtile = wave >> 3;        // 8 n-tiles cover the 128 fused output cols
  int ntile = wave & 7;
  if (mtile >= mtiles) return;  // wave-uniform: EXEC stays all-ones for WMMA

  int m0 = mtile * 16;
  int colg = ntile * 16 + (l & 15);
  int rbase = m0 + ((l & 16) ? 8 : 0);
  int arow = m0 + (l & 15);
  if (arow >= N) arow = N - 1;  // per-lane clamp (data only, no divergence)

  const _Float16* brow = wsw + (((size_t)ntile * 8) * 32 + l) * 16;

  v8f c = {};
#pragma unroll
  for (int ki = 0; ki < 8; ++ki) {
    int k0 = ki * 32;
    // A 16x32: lanes 0-15 hold K[k0..k0+15]; lanes 16-31 hold K[k0+16..k0+31]
    int kb = k0 + ((l & 16) ? 16 : 0);
    const _Float16* ap = h16 + (size_t)arow * 256 + kb;
    frag16 a, b;
    a.h[0] = *(const v8h*)ap;
    a.h[1] = *(const v8h*)(ap + 8);
    // B fragment: 32 contiguous bytes in the swizzled weight buffer
    const _Float16* bp = brow + (size_t)ki * (32 * 16);
    b.h[0] = *(const v8h*)bp;
    b.h[1] = *(const v8h*)(bp + 8);
    c = __builtin_amdgcn_wmma_f32_16x16x32_f16(false, a.v, false, b.v,
                                               (short)0, c, false, false);
  }
  float bias = (colg < 64) ? b2l[colg] : b2r[colg - 64];
#pragma unroll
  for (int r = 0; r < 8; ++r) {
    int row = rbase + r;
    if (row < N) {
      float v = c[r] + bias;
      if (colg < 64) xl2h[(size_t)row * 64 + colg] = (_Float16)v;
      else           xr2h[(size_t)row * 64 + (colg - 64)] = (_Float16)v;
    }
  }
}

// ---- conv2 edge phase (1 head x 64ch): wave per edge, 2 ch/lane ------------

__global__ void __launch_bounds__(256) k_att2_logits(
    const int* __restrict__ eidx, const float* __restrict__ eattr,
    const float* __restrict__ loopattr,
    const _Float16* __restrict__ xl2h, const _Float16* __restrict__ xr2h,
    const float* __restrict__ W2e, const float* __restrict__ att2,
    float* __restrict__ alpha2, unsigned* __restrict__ amax2, int E, int EP) {
  int e = blockIdx.x * 8 + (threadIdx.x >> 5);
  if (e >= EP) return;
  int l = threadIdx.x & 31;
  int src, dst; float ea0, ea1;
  if (e < E) {
    src = eidx[e]; dst = eidx[(size_t)E + e];
    ea0 = eattr[2 * (size_t)e]; ea1 = eattr[2 * (size_t)e + 1];
  } else {
    int n = e - E; src = n; dst = n;
    ea0 = loopattr[2 * (size_t)n]; ea1 = loopattr[2 * (size_t)n + 1];
  }
  int c0 = l * 2;
  v2h xlv = *(const v2h*)(xl2h + (size_t)src * 64 + c0);
  v2h xrv = *(const v2h*)(xr2h + (size_t)dst * 64 + c0);
  float p = 0.0f;
#pragma unroll
  for (int i = 0; i < 2; ++i) {
    int j = c0 + i;
    float v = (float)xlv[i] + (float)xrv[i] + ea0 * W2e[j] + ea1 * W2e[64 + j];
    v = (v > 0.0f) ? v : 0.2f * v;
    p += v * att2[j];
  }
  p += __shfl_xor(p, 1, 32);
  p += __shfl_xor(p, 2, 32);
  p += __shfl_xor(p, 4, 32);
  p += __shfl_xor(p, 8, 32);
  p += __shfl_xor(p, 16, 32);
  if (l == 0) {
    alpha2[e] = p;
    atomicMax(amax2 + dst, encOrd(p));
  }
}

__global__ void k_att2_exp(const int* __restrict__ eidx, float* __restrict__ alpha2,
                           const unsigned* __restrict__ amax2, float* __restrict__ denom2,
                           int E, int EP) {
  int e = blockIdx.x * blockDim.x + threadIdx.x;
  if (e >= EP) return;
  int dst = (e < E) ? eidx[(size_t)E + e] : (e - E);
  float ex = __expf(alpha2[e] - decOrd(amax2[dst]));
  alpha2[e] = ex;
  atomicAdd(denom2 + dst, ex);
}

__global__ void __launch_bounds__(256) k_att2_aggr(
    const int* __restrict__ eidx, const float* __restrict__ alpha2,
    const float* __restrict__ denom2, const _Float16* __restrict__ xl2h,
    float* __restrict__ acc2, int E, int EP) {
  int e = blockIdx.x * 8 + (threadIdx.x >> 5);
  if (e >= EP) return;
  int l = threadIdx.x & 31;
  int src, dst;
  if (e < E) { src = eidx[e]; dst = eidx[(size_t)E + e]; }
  else       { src = dst = e - E; }
  float w = alpha2[e] / (denom2[dst] + 1e-16f);
  int c0 = l * 2;
  v2h xlv = *(const v2h*)(xl2h + (size_t)src * 64 + c0);
  float* accp = acc2 + (size_t)dst * 64 + c0;
  atomicAdd(accp,     (float)xlv[0] * w);
  atomicAdd(accp + 1, (float)xlv[1] * w);
}

// ---- mean pool -------------------------------------------------------------

__global__ void k_bias_out(const float* __restrict__ bias2, float* __restrict__ out) {
  int j = threadIdx.x;
  if (j < 64) out[j] = bias2[j];   // mean(out + bias2) == mean(out) + bias2
}

__global__ void __launch_bounds__(256) k_pool(const float* __restrict__ acc2,
                                              float* __restrict__ out, int N) {
  int c  = threadIdx.x & 63;
  int rg = threadIdx.x >> 6;        // 4 node-rows per block per iteration
  float s = 0.0f;
  for (int n = blockIdx.x * 4 + rg; n < N; n += gridDim.x * 4)
    s += acc2[(size_t)n * 64 + c];
  atomicAdd(out + c, s * (1.0f / (float)N));
}

// ---------------------------------------------------------------------------

extern "C" void kernel_launch(void* const* d_in, const int* in_sizes, int n_in,
                              void* d_out, int out_size, void* d_ws, size_t ws_size,
                              hipStream_t stream) {
  const float* x     = (const float*)d_in[0];
  const float* eattr = (const float*)d_in[1];
  const float* W1l   = (const float*)d_in[2];
  const float* b1l   = (const float*)d_in[3];
  const float* W1r   = (const float*)d_in[4];
  const float* b1r   = (const float*)d_in[5];
  const float* W1e   = (const float*)d_in[6];
  const float* att1  = (const float*)d_in[7];
  const float* bias1 = (const float*)d_in[8];
  const float* W2l   = (const float*)d_in[9];
  const float* b2l   = (const float*)d_in[10];
  const float* W2r   = (const float*)d_in[11];
  const float* b2r   = (const float*)d_in[12];
  const float* W2e   = (const float*)d_in[13];
  const float* att2  = (const float*)d_in[14];
  const float* bias2 = (const float*)d_in[15];
  const int*   eidx  = (const int*)d_in[16];
  float* out = (float*)d_out;

  const int N  = in_sizes[0] / 2;    // 50000 (x is [N,2])
  const int E  = in_sizes[16] / 2;   // 500000
  const int EP = E + N;              // edges + self loops

  // ---- carve workspace: zero-init accumulators first, then raw scratch ----
  char* base = (char*)d_ws;
  size_t off = 0;
  auto take = [&](size_t bytes) -> void* {
    void* p = base + off;
    off += (bytes + 255) & ~(size_t)255;
    return p;
  };
  float*    deg      = (float*)   take((size_t)N * 4);
  float*    loopattr = (float*)   take((size_t)N * 2 * 4);
  unsigned* amax1    = (unsigned*)take((size_t)N * 4 * 4);   // 0 == "-inf" in encOrd order
  float*    denom1   = (float*)   take((size_t)N * 4 * 4);
  unsigned* amax2    = (unsigned*)take((size_t)N * 4);
  float*    denom2   = (float*)   take((size_t)N * 4);
  float*    acc1     = (float*)   take((size_t)N * 256 * 4);
  float*    acc2     = (float*)   take((size_t)N * 64 * 4);
  const size_t zero_bytes = off;                             // prefix to memset
  float*    alpha1   = (float*)   take((size_t)EP * 4 * 4);
  float*    alpha2   = (float*)   take((size_t)EP * 4);
  _Float16* xl1h     = (_Float16*)take((size_t)N * 256 * 2);
  _Float16* xr1h     = (_Float16*)take((size_t)N * 256 * 2);
  _Float16* h16      = (_Float16*)take((size_t)N * 256 * 2);
  _Float16* wsw      = (_Float16*)take((size_t)256 * 128 * 2);
  _Float16* xl2h     = (_Float16*)take((size_t)N * 64 * 2);
  _Float16* xr2h     = (_Float16*)take((size_t)N * 64 * 2);
  (void)ws_size; (void)n_in; (void)out_size;

  hipMemsetAsync(d_ws, 0, zero_bytes, stream);

  // conv1
  k_deg<<<(E + 255) / 256, 256, 0, stream>>>(eidx, eattr, deg, loopattr, E);
  k_loopdiv<<<(N + 255) / 256, 256, 0, stream>>>(loopattr, deg, N);
  {
    size_t tot = (size_t)N * 256;
    k_node1<<<(unsigned)((tot + 255) / 256), 256, 0, stream>>>(x, W1l, b1l, W1r, b1r,
                                                               xl1h, xr1h, N);
  }
  k_att1_logits<<<(EP + 7) / 8, 256, 0, stream>>>(eidx, eattr, loopattr, xl1h, xr1h,
                                                  W1e, att1, alpha1, amax1, E, EP);
  k_att1_exp<<<(EP * 4 + 255) / 256, 256, 0, stream>>>(eidx, alpha1, amax1, denom1, E, EP);
  k_att1_aggr<<<(EP + 7) / 8, 256, 0, stream>>>(eidx, alpha1, denom1, xl1h, acc1, E, EP);
  {
    size_t tot = (size_t)N * 256;
    k_finish1<<<(unsigned)((tot + 255) / 256), 256, 0, stream>>>(acc1, bias1, h16, N);
  }

  // conv2 node transform via WMMA (B pre-swizzled to fragment layout)
  k_wcat_sw<<<(256 * 128 + 255) / 256, 256, 0, stream>>>(W2l, W2r, wsw);
  {
    int mtiles = (N + 15) / 16;            // 3125
    int waves  = mtiles * 8;               // 8 n-tiles (128 fused cols)
    k_gemm2<<<(waves + 3) / 4, 128, 0, stream>>>(h16, wsw, b2l, b2r, xl2h, xr2h, N, mtiles);
  }

  // conv2 edge phase
  k_att2_logits<<<(EP + 7) / 8, 256, 0, stream>>>(eidx, eattr, loopattr, xl2h, xr2h,
                                                  W2e, att2, alpha2, amax2, E, EP);
  k_att2_exp<<<(EP + 255) / 256, 256, 0, stream>>>(eidx, alpha2, amax2, denom2, E, EP);
  k_att2_aggr<<<(EP + 7) / 8, 256, 0, stream>>>(eidx, alpha2, denom2, xl2h, acc2, E, EP);

  // global mean pool (+bias2)
  k_bias_out<<<1, 64, 0, stream>>>(bias2, out);
  k_pool<<<256, 256, 0, stream>>>(acc2, out, N);
}
